// SimCLRContrastiveLoss_19799799234826
// MI455X (gfx1250) — compile-verified, matrix operand order checked by reading, not run on hardware
//
#include <hip/hip_runtime.h>
#include <hip/hip_bf16.h>

typedef __attribute__((ext_vector_type(16))) _Float16 v16h;
typedef __attribute__((ext_vector_type(8)))  float    v8f;

#define NROWS 8192
#define DDIM  256
#define INV_TEMP 2.0f

union F16Frag { v16h v; uint4 u[2]; };
struct alignas(16) H8 { _Float16 h[8]; };

// ---------------------------------------------------------------------------
// Kernel 1: L2-normalize rows of z1 and z2, emit f16 rows into workspace.
// One wave (32 lanes) per row; 8 floats per lane.
// ---------------------------------------------------------------------------
__global__ void __launch_bounds__(256)
normalize_f16_kernel(const float* __restrict__ z1, const float* __restrict__ z2,
                     _Float16* __restrict__ z1h, _Float16* __restrict__ z2h) {
  const int wave = threadIdx.x >> 5;
  const int lane = threadIdx.x & 31;
  const int row  = blockIdx.x * 8 + wave;   // 0 .. 2*NROWS-1

  const float* src;
  _Float16*    dst;
  if (row < NROWS) { src = z1 + (size_t)row * DDIM;           dst = z1h + (size_t)row * DDIM; }
  else             { src = z2 + (size_t)(row - NROWS) * DDIM; dst = z2h + (size_t)(row - NROWS) * DDIM; }

  const float4* srcv = reinterpret_cast<const float4*>(src);
  float4 a = srcv[lane * 2 + 0];
  float4 b = srcv[lane * 2 + 1];

  float ss = a.x*a.x + a.y*a.y + a.z*a.z + a.w*a.w
           + b.x*b.x + b.y*b.y + b.z*b.z + b.w*b.w;
#pragma unroll
  for (int m = 1; m <= 16; m <<= 1) ss += __shfl_xor(ss, m, 32);

  float norm  = sqrtf(ss);
  float scale = 1.0f / fmaxf(norm, 1e-12f);

  H8 o;
  o.h[0] = (_Float16)(a.x * scale);
  o.h[1] = (_Float16)(a.y * scale);
  o.h[2] = (_Float16)(a.z * scale);
  o.h[3] = (_Float16)(a.w * scale);
  o.h[4] = (_Float16)(b.x * scale);
  o.h[5] = (_Float16)(b.y * scale);
  o.h[6] = (_Float16)(b.z * scale);
  o.h[7] = (_Float16)(b.w * scale);
  reinterpret_cast<H8*>(dst)[lane] = o;
}

// ---------------------------------------------------------------------------
// Kernel 2: fused scores-GEMM + online logsumexp + diagonal capture.
// Grid: 64 blocks x 256 threads (8 waves). Each wave owns 16 rows and streams
// all 8192 columns through v_wmma_f32_16x16x32_f16, keeping per-row
// (running max, running sumexp) replicated across its half-wave lanes.
// ---------------------------------------------------------------------------
__global__ void __launch_bounds__(256)
simclr_lse_kernel(const _Float16* __restrict__ z1h,
                  const _Float16* __restrict__ z2h,
                  float* __restrict__ rowloss) {
  const int lane = threadIdx.x & 31;
  const int wave = threadIdx.x >> 5;
  const int hi   = (lane >> 4) & 1;   // half-wave selector
  const int lm   = lane & 15;
  const int rowBase = blockIdx.x * 128 + wave * 16;

  // --- Preload A fragments (16 rows x 256 cols) per ISA 16-bit A layout:
  // lanes 0-15 hold row M=lm, K chunks {k+0..7, k+16..23};
  // lanes 16-31 hold row M=lm, K chunks {k+8..15, k+24..31}.
  F16Frag afrag[8];
  const _Float16* arow = z1h + (size_t)(rowBase + lm) * DDIM;
#pragma unroll
  for (int kk = 0; kk < 8; ++kk) {
    const int c0 = kk * 32 + hi * 8;
    afrag[kk].u[0] = *reinterpret_cast<const uint4*>(arow + c0);
    afrag[kk].u[1] = *reinterpret_cast<const uint4*>(arow + c0 + 16);
  }

  float runMax[8], runSum[8], diag[8];
#pragma unroll
  for (int r = 0; r < 8; ++r) { runMax[r] = -INFINITY; runSum[r] = 0.0f; diag[r] = 0.0f; }

  const int absRow0 = rowBase + hi * 8;  // C/D layout: VGPR r holds rows r (lanes 0-15) and r+8 (lanes 16-31)

  for (int colBase = 0; colBase < NROWS; colBase += 64) {
    v8f acc[4];
#pragma unroll
    for (int t = 0; t < 4; ++t) {
      v8f c = {};
      const _Float16* brow = z2h + (size_t)(colBase + t * 16 + lm) * DDIM;
#pragma unroll
      for (int kk = 0; kk < 8; ++kk) {
        F16Frag bf;
        const int c0 = kk * 32 + hi * 8;
        bf.u[0] = *reinterpret_cast<const uint4*>(brow + c0);
        bf.u[1] = *reinterpret_cast<const uint4*>(brow + c0 + 16);
        c = __builtin_amdgcn_wmma_f32_16x16x32_f16(
            /*neg_a=*/false, afrag[kk].v, /*neg_b=*/false, bf.v,
            /*c_mod=*/(short)0, c, /*reuse_a=*/false, /*reuse_b=*/false);
      }
      acc[t] = c;
    }

    // Online softmax update for the 64 new columns.
#pragma unroll
    for (int r = 0; r < 8; ++r) {
      float v0 = acc[0][r] * INV_TEMP;
      float v1 = acc[1][r] * INV_TEMP;
      float v2 = acc[2][r] * INV_TEMP;
      float v3 = acc[3][r] * INV_TEMP;

      const int absRow = absRow0 + r;
      const int col0   = colBase + lm;
      if (col0      == absRow) diag[r] = v0;
      if (col0 + 16 == absRow) diag[r] = v1;
      if (col0 + 32 == absRow) diag[r] = v2;
      if (col0 + 48 == absRow) diag[r] = v3;

      float tm = fmaxf(fmaxf(v0, v1), fmaxf(v2, v3));
#pragma unroll
      for (int m = 1; m <= 8; m <<= 1) tm = fmaxf(tm, __shfl_xor(tm, m, 32));

      const float nm = fmaxf(runMax[r], tm);
      float s = __expf(v0 - nm) + __expf(v1 - nm) + __expf(v2 - nm) + __expf(v3 - nm);
#pragma unroll
      for (int m = 1; m <= 8; m <<= 1) s += __shfl_xor(s, m, 32);

      runSum[r] = runSum[r] * __expf(runMax[r] - nm) + s;
      runMax[r] = nm;
    }
  }

  // Emit per-row loss: lse - diag. Diagonal lives in exactly one lane of the
  // half-wave; a half-wave sum broadcast recovers it everywhere.
#pragma unroll
  for (int r = 0; r < 8; ++r) {
    float d = diag[r];
#pragma unroll
    for (int m = 1; m <= 8; m <<= 1) d += __shfl_xor(d, m, 32);
    if (lm == 0) {
      rowloss[absRow0 + r] = (__logf(runSum[r]) + runMax[r]) - d;
    }
  }
}

// ---------------------------------------------------------------------------
// Kernel 3: mean over 8192 per-row losses -> scalar.
// ---------------------------------------------------------------------------
__global__ void __launch_bounds__(256)
reduce_loss_kernel(const float* __restrict__ rowloss, float* __restrict__ out) {
  __shared__ float sm[256];
  float s = 0.0f;
  for (int i = threadIdx.x; i < NROWS; i += 256) s += rowloss[i];
  sm[threadIdx.x] = s;
  __syncthreads();
  for (int off = 128; off > 0; off >>= 1) {
    if ((int)threadIdx.x < off) sm[threadIdx.x] += sm[threadIdx.x + off];
    __syncthreads();
  }
  if (threadIdx.x == 0) out[0] = sm[0] / (float)NROWS;
}

// ---------------------------------------------------------------------------
extern "C" void kernel_launch(void* const* d_in, const int* in_sizes, int n_in,
                              void* d_out, int out_size, void* d_ws, size_t ws_size,
                              hipStream_t stream) {
  const float* z1 = (const float*)d_in[0];
  const float* z2 = (const float*)d_in[1];
  float* out = (float*)d_out;

  // Workspace layout: z1h (4 MB) | z2h (4 MB) | rowloss (32 KB)
  char* ws = (char*)d_ws;
  _Float16* z1h = (_Float16*)(ws);
  _Float16* z2h = (_Float16*)(ws + (size_t)NROWS * DDIM * sizeof(_Float16));
  float* rowloss = (float*)(ws + 2 * (size_t)NROWS * DDIM * sizeof(_Float16));

  // 1) normalize both matrices into f16 (16384 rows, 8 rows/block)
  normalize_f16_kernel<<<(2 * NROWS) / 8, 256, 0, stream>>>(z1, z2, z1h, z2h);

  // 2) fused GEMM + online LSE + diagonal (64 blocks x 8 waves x 16 rows)
  simclr_lse_kernel<<<NROWS / 128, 256, 0, stream>>>(z1h, z2h, rowloss);

  // 3) scalar mean
  reduce_loss_kernel<<<1, 256, 0, stream>>>(rowloss, out);
}
